// TopologyAwareGSSPoolingLayer_42219528520316
// MI455X (gfx1250) — matrix operand order repaired and checked by reading, not compiled, not run on hardware
//
#include <hip/hip_runtime.h>
#include <math.h>

#define NN 2048
#define DD 512
#define SS 1024   // POOLING_RATIO * NN

typedef __attribute__((ext_vector_type(16))) _Float16 v16h;
typedef __attribute__((ext_vector_type(8)))  _Float16 v8h;
typedef __attribute__((ext_vector_type(8)))  float    v8f;

// ---------------------------------------------------------------------------
// K0: deg = adj.sum(1); adjH = f16(adj); lapH = f16(diag(deg) - adj)
// ---------------------------------------------------------------------------
__global__ void k_prep(const float* __restrict__ adj, _Float16* __restrict__ adjH,
                       _Float16* __restrict__ lapH, float* __restrict__ deg) {
  int row = blockIdx.x;
  int tid = threadIdx.x;                 // 256 threads
  __shared__ float red[256];
  float s = 0.f;
  for (int j = tid; j < NN; j += 256) s += adj[(size_t)row * NN + j];
  red[tid] = s; __syncthreads();
  for (int o = 128; o > 0; o >>= 1) { if (tid < o) red[tid] += red[tid + o]; __syncthreads(); }
  float d = red[0];
  if (tid == 0) deg[row] = d;
  for (int j = tid; j < NN; j += 256) {
    float a = adj[(size_t)row * NN + j];
    adjH[(size_t)row * NN + j] = (_Float16)a;
    lapH[(size_t)row * NN + j] = (_Float16)((j == row) ? d : -a);
  }
}

__global__ void k_f32_to_f16(const float* __restrict__ in, _Float16* __restrict__ out, int n) {
  int i = blockIdx.x * blockDim.x + threadIdx.x;
  if (i < n) out[i] = (_Float16)in[i];
}

// ---------------------------------------------------------------------------
// WMMA GEMM: C[M,N] (f32) = A[M,K] @ B[K,N], both f16 row-major.
// One wave per 16x64 C slab: A fragment loaded once, 4 WMMAs per K-step
// (4 independent accumulator chains hide the f16-WMMA RAW latency).
// Block = 128 threads = 4 waves -> 16x256 per block. M%16==0, N%64==0, K%32==0.
// ---------------------------------------------------------------------------
__global__ void __launch_bounds__(128)
k_wmma_gemm(const _Float16* __restrict__ A, const _Float16* __restrict__ B,
            float* __restrict__ C, int M, int N, int K) {
  int wave = threadIdx.x >> 5;
  int lane = threadIdx.x & 31;
  int tn = (blockIdx.x * 4 + wave) * 64;   // N slab origin (64 wide)
  int tm = blockIdx.y * 16;                // M tile origin
  if (tn >= N || tm >= M) return;          // wave-uniform

  int half = lane >> 4;                    // 0: K-lo chunks, 1: K-hi chunks
  int mrow = tm + (lane & 15);             // A row held by this lane
  v8f acc0 = {}, acc1 = {}, acc2 = {}, acc3 = {};

  for (int k0 = 0; k0 < K; k0 += 32) {
    // --- A tile (16x32): lane holds row mrow, two contiguous 8xf16 K-chunks
    const _Float16* Ap = A + (size_t)mrow * K + k0 + half * 8;
    v8h alo = *(const v8h*)(Ap);           // K = k0+half*8 .. +7
    v8h ahi = *(const v8h*)(Ap + 16);      // K = k0+half*8+16 .. +23
    v16h a;
#pragma unroll
    for (int h = 0; h < 8; ++h) { a[h] = alo[h]; a[h + 8] = ahi[h]; }

    // --- B tiles (32x16 x4): lane holds column tn+(lane&15), K in same order as A
    int kb = k0;
    const _Float16* Bp = B + (size_t)kb * N + tn;
    v16h b0, b1, b2, b3;
#pragma unroll
    for (int s = 0; s < 16; ++s) {
      int kk = (s < 8) ? (half * 8 + s) : (16 + half * 8 + (s - 8));
      const _Float16* q = Bp + (size_t)kk * N + (lane & 15);
      b0[s] = q[0]; b1[s] = q[16]; b2[s] = q[32]; b3[s] = q[48];
    }

    if (k0 + 32 < K) {                     // global_prefetch_b8 path
      __builtin_prefetch(Ap + 32, 0, 1);
      __builtin_prefetch(Bp + (size_t)32 * N, 0, 1);
    }

    acc0 = __builtin_amdgcn_wmma_f32_16x16x32_f16(false, a, false, b0, (short)0, acc0, false, false);
    acc1 = __builtin_amdgcn_wmma_f32_16x16x32_f16(false, a, false, b1, (short)0, acc1, false, false);
    acc2 = __builtin_amdgcn_wmma_f32_16x16x32_f16(false, a, false, b2, (short)0, acc2, false, false);
    acc3 = __builtin_amdgcn_wmma_f32_16x16x32_f16(false, a, false, b3, (short)0, acc3, false, false);
  }

  // C layout: VGPR r -> M = tm + r + 8*half, N = tn + (lane&15) (+16 per tile)
  int cn = tn + (lane & 15);
#pragma unroll
  for (int r = 0; r < 8; ++r) {
    int cm = tm + r + half * 8;
    float* Cp = C + (size_t)cm * N + cn;
    Cp[0]  = acc0[r];
    Cp[16] = acc1[r];
    Cp[32] = acc2[r];
    Cp[48] = acc3[r];
  }
}

// ---------------------------------------------------------------------------
// K2: v[row] = || y2[row, :] ||_2
// ---------------------------------------------------------------------------
__global__ void k_rownorm(const float* __restrict__ y, float* __restrict__ v) {
  int row = blockIdx.x;
  int tid = threadIdx.x;                   // 256
  __shared__ float red[256];
  float s = 0.f;
  for (int j = tid; j < DD; j += 256) { float t = y[(size_t)row * DD + j]; s += t * t; }
  red[tid] = s; __syncthreads();
  for (int o = 128; o > 0; o >>= 1) { if (tid < o) red[tid] += red[tid + o]; __syncthreads(); }
  if (tid == 0) v[row] = sqrtf(red[0]);
}

// ---------------------------------------------------------------------------
// K3: ab[i,j] = 1 iff (adj+adj2)==0 else 0 (int8); initial per-row argmax of
// sm[i,j] = v_i * ab_ij * v_j, tie -> lowest column.
// ---------------------------------------------------------------------------
__global__ void k_abrow(const float* __restrict__ adj, const float* __restrict__ adj2,
                        const float* __restrict__ v, signed char* __restrict__ ab,
                        float* __restrict__ rowmax, int* __restrict__ rowarg) {
  int row = blockIdx.x;
  int tid = threadIdx.x;                   // 256
  __shared__ float rv[256];
  __shared__ int   ri[256];
  float vi = v[row];
  float best = -2.f; int bi = NN;
  for (int j = tid; j < NN; j += 256) {
    float acc = adj[(size_t)row * NN + j] + adj2[(size_t)row * NN + j];
    signed char m = (acc < 0.5f) ? 1 : 0;
    ab[(size_t)row * NN + j] = m;
    float s = m ? vi * v[j] : 0.f;
    if (s > best || (s == best && j < bi)) { best = s; bi = j; }
  }
  rv[tid] = best; ri[tid] = bi; __syncthreads();
  for (int o = 128; o > 0; o >>= 1) {
    if (tid < o) {
      if (rv[tid + o] > rv[tid] || (rv[tid + o] == rv[tid] && ri[tid + o] < ri[tid])) {
        rv[tid] = rv[tid + o]; ri[tid] = ri[tid + o];
      }
    }
    __syncthreads();
  }
  if (tid == 0) { rowmax[row] = rv[0]; rowarg[row] = ri[0]; }
}

// ---------------------------------------------------------------------------
// K4: greedy sampler, single workgroup (1024 threads), row-max caching.
// ab codes: 1 edge, 0 no-edge, 2 edge+suppressed, -1 no-edge+suppressed.
// sm value: 1 -> v_i*v_j, 0 -> 0, 2/-1 -> -1.
// ---------------------------------------------------------------------------
__global__ void __launch_bounds__(1024)
k_sampler(volatile signed char* ab, const float* __restrict__ v,
          const float* __restrict__ rowmaxg, const int* __restrict__ rowargg,
          int* __restrict__ idx_out, float* __restrict__ valid_out) {
  __shared__ float s_rowmax[NN];
  __shared__ int   s_rowarg[NN];
  __shared__ float s_v[NN];
  __shared__ unsigned char s_mask[NN];
  __shared__ float s_rval[1024];
  __shared__ int   s_ridx[1024];
  __shared__ int s_i0, s_i1, s_done, s_sup, s_bad, s_count;

  int tid = threadIdx.x;
  for (int r = tid; r < NN; r += 1024) {
    s_rowmax[r] = rowmaxg[r]; s_rowarg[r] = rowargg[r];
    s_v[r] = v[r]; s_mask[r] = 0;
  }
  if (tid == 0) { s_done = 0; s_count = 0; s_sup = 0; }
  __syncthreads();

  for (int iter = 0; iter < NN; ++iter) {
    // ---- global argmax over row maxima (tie -> lowest row) ----
    float b0 = s_rowmax[tid]; int r0 = tid;
    float b1 = s_rowmax[tid + 1024];
    if (b1 > b0) { b0 = b1; r0 = tid + 1024; }
    s_rval[tid] = b0; s_ridx[tid] = r0;
    __syncthreads();
    for (int o = 512; o > 0; o >>= 1) {
      if (tid < o) {
        float ov = s_rval[tid + o]; int oi = s_ridx[tid + o];
        if (ov > s_rval[tid] || (ov == s_rval[tid] && oi < s_ridx[tid])) {
          s_rval[tid] = ov; s_ridx[tid] = oi;
        }
      }
      __syncthreads();
    }
    if (tid == 0) { s_i0 = s_ridx[0]; s_i1 = s_rowarg[s_ridx[0]]; s_bad = 0; }
    __syncthreads();
    int i0 = s_i0, i1 = s_i1;
    int is_first = (s_count == 0);         // uniform

    if (is_first) {
      if (tid == 0) {
        s_mask[i0] = 1; s_mask[i1] = 1;
        s_count = (i0 == i1) ? 1 : 2;
        s_sup = 1;                          // suppress; done stays 0
      }
    } else {
      // ok0: all sampled rows (old mask) have ab[:,i0] != 0
      for (int r = tid; r < NN; r += 1024) {
        if (s_mask[r]) { signed char m = ab[(size_t)r * NN + i0]; if (!(m == 1 || m == 2)) s_bad = 1; }
      }
      __syncthreads();
      if (tid == 0) {
        int ok0 = !s_bad;
        if (ok0 && !s_mask[i0]) { s_mask[i0] = 1; s_count++; }
        s_bad = 0;
        s_done = (s_count == SS);           // done1
      }
      __syncthreads();
      int done1 = s_done;                   // uniform
      if (!done1) {
        for (int r = tid; r < NN; r += 1024) {
          if (s_mask[r]) { signed char m = ab[(size_t)r * NN + i1]; if (!(m == 1 || m == 2)) s_bad = 1; }
        }
      }
      __syncthreads();
      if (tid == 0) {
        int ok1 = (!done1) && (!s_bad);
        if (ok1 && !s_mask[i1]) { s_mask[i1] = 1; s_count++; }
        int done2 = (s_count == SS);
        s_done = done2;
        s_sup  = !done2;                    // suppress only if not breaking
      }
    }
    __syncthreads();
    if (s_done) break;

    if (s_sup) {
      if (tid == 0) {
        signed char m01 = ab[(size_t)i0 * NN + i1];
        ab[(size_t)i0 * NN + i1] = (m01 == 1 || m01 == 2) ? (signed char)2 : (signed char)-1;
        signed char m10 = ab[(size_t)i1 * NN + i0];
        ab[(size_t)i1 * NN + i0] = (m10 == 1 || m10 == 2) ? (signed char)2 : (signed char)-1;
      }
      __threadfence();
      __syncthreads();
      // recompute row maxima for the two touched rows only
      for (int pass = 0; pass < 2; ++pass) {
        if (pass == 1 && i1 == i0) break;   // uniform
        int R    = pass ? i1 : i0;
        int skip = pass ? i0 : i1;
        float vR = s_v[R];
        float best = -2.f; int bi = NN;
        for (int c = tid; c < NN; c += 1024) {
          signed char m = ab[(size_t)R * NN + c];
          float s = (m == 1) ? vR * s_v[c] : ((m == 0) ? 0.f : -1.f);
          if (c == skip) s = -1.f;          // belt & braces vs store visibility
          if (s > best || (s == best && c < bi)) { best = s; bi = c; }
        }
        s_rval[tid] = best; s_ridx[tid] = bi;
        __syncthreads();
        for (int o = 512; o > 0; o >>= 1) {
          if (tid < o) {
            float ov = s_rval[tid + o]; int oi = s_ridx[tid + o];
            if (ov > s_rval[tid] || (ov == s_rval[tid] && oi < s_ridx[tid])) {
              s_rval[tid] = ov; s_ridx[tid] = oi;
            }
          }
          __syncthreads();
        }
        if (tid == 0) { s_rowmax[R] = s_rval[0]; s_rowarg[R] = s_ridx[0]; }
        __syncthreads();
      }
    }
    __syncthreads();
  }
  __syncthreads();

  if (tid == 0) {                           // emit idx (ascending) + valid
    int pos = 0;
    for (int r = 0; r < NN; ++r)
      if (s_mask[r] && pos < SS) idx_out[pos++] = r;
    int total = pos;
    for (int p = pos; p < SS; ++p) idx_out[p] = 0;
    for (int p = 0; p < SS; ++p) valid_out[p] = (p < total) ? 1.f : 0.f;
  }
}

// ---------------------------------------------------------------------------
// K5: pooling
// ---------------------------------------------------------------------------
__global__ void k_newx(const float* __restrict__ x, const float* __restrict__ ax,
                       const int* __restrict__ idx, const float* __restrict__ validf,
                       float* __restrict__ outx) {
  int r = blockIdx.x;
  int tid = threadIdx.x;                    // 256
  int i = idx[r];
  float vf = validf[r];
  for (int c = tid; c < DD; c += 256)
    outx[(size_t)r * DD + c] = vf * (x[(size_t)i * DD + c] + ax[(size_t)i * DD + c]);
}

__global__ void k_newadj(const float* __restrict__ adj2, const int* __restrict__ idx,
                         const float* __restrict__ validf, float* __restrict__ outadj) {
  int g = blockIdx.x * blockDim.x + threadIdx.x;
  if (g >= SS * SS) return;
  int r = g / SS, c = g % SS;
  int i = idx[r], j = idx[c];
  float s = adj2[(size_t)i * NN + j];
  float t = (s > 0.f ? 1.f : 0.f) - ((i == j) ? s : 0.f);
  outadj[g] = t * validf[r] * validf[c];
}

// ---------------------------------------------------------------------------
extern "C" void kernel_launch(void* const* d_in, const int* in_sizes, int n_in,
                              void* d_out, int out_size, void* d_ws, size_t ws_size,
                              hipStream_t stream) {
  const float* x   = (const float*)d_in[0];   // [NN, DD]
  const float* adj = (const float*)d_in[1];   // [NN, NN]

  char* ws = (char*)d_ws;
  size_t off = 0;
  auto take = [&](size_t bytes) { size_t o = off; off += (bytes + 255) & ~(size_t)255; return o; };

  _Float16* adjH  = (_Float16*)(ws + take((size_t)NN * NN * 2));
  _Float16* lapH  = (_Float16*)(ws + take((size_t)NN * NN * 2));
  _Float16* xH    = (_Float16*)(ws + take((size_t)NN * DD * 2));
  float*    adj2  = (float*)   (ws + take((size_t)NN * NN * 4));
  signed char* ab = (signed char*)(ws + take((size_t)NN * NN));
  float*    y32   = (float*)   (ws + take((size_t)NN * DD * 4)); // y1 then y2
  _Float16* yH    = (_Float16*)(ws + take((size_t)NN * DD * 2));
  float*    ax    = (float*)   (ws + take((size_t)NN * DD * 4));
  float*    vbuf  = (float*)   (ws + take((size_t)NN * 4));
  float*    rowmx = (float*)   (ws + take((size_t)NN * 4));
  int*      rowar = (int*)     (ws + take((size_t)NN * 4));
  float*    degb  = (float*)   (ws + take((size_t)NN * 4));
  int*      idxb  = (int*)     (ws + take((size_t)SS * 4));
  float*    valb  = (float*)   (ws + take((size_t)SS * 4));
  (void)ws_size; (void)in_sizes; (void)n_in; (void)out_size;

  float* outx   = (float*)d_out;             // [SS, DD]
  float* outadj = (float*)d_out + (size_t)SS * DD; // [SS, SS]

  // prep: deg, adjH, lapH; xH
  k_prep<<<NN, 256, 0, stream>>>(adj, adjH, lapH, degb);
  k_f32_to_f16<<<(NN * DD + 255) / 256, 256, 0, stream>>>(x, xH, NN * DD);

  // adj2 = adj @ adj   (WMMA, M=N=K=2048)
  k_wmma_gemm<<<dim3(NN / 256, NN / 16), 128, 0, stream>>>(adjH, adjH, adj2, NN, NN, NN);

  // y1 = lap @ x ; y2 = lap @ y1  (variation)
  k_wmma_gemm<<<dim3(DD / 256, NN / 16), 128, 0, stream>>>(lapH, xH, y32, NN, DD, NN);
  k_f32_to_f16<<<(NN * DD + 255) / 256, 256, 0, stream>>>(y32, yH, NN * DD);
  k_wmma_gemm<<<dim3(DD / 256, NN / 16), 128, 0, stream>>>(lapH, yH, y32, NN, DD, NN);
  k_rownorm<<<NN, 256, 0, stream>>>(y32, vbuf);

  // ax = adj @ x (needed for pooling, independent of sampler)
  k_wmma_gemm<<<dim3(DD / 256, NN / 16), 128, 0, stream>>>(adjH, xH, ax, NN, DD, NN);

  // adj_bar (int8) + initial per-row maxima of sm
  k_abrow<<<NN, 256, 0, stream>>>(adj, adj2, vbuf, ab, rowmx, rowar);

  // greedy sampler (single workgroup, row-max cached)
  k_sampler<<<1, 1024, 0, stream>>>(ab, vbuf, rowmx, rowar, idxb, valb);

  // pooled outputs
  k_newx<<<SS, 256, 0, stream>>>(x, ax, idxb, valb, outx);
  k_newadj<<<(SS * SS + 255) / 256, 256, 0, stream>>>(adj2, idxb, valb, outadj);
}
